// Encoder_51582557225688
// MI455X (gfx1250) — matrix operand order repaired and checked by reading, not compiled, not run on hardware
//
#include <hip/hip_runtime.h>

// ============================================================================
// CDNA5 (gfx1250) 5-stream memory-transformer encoder.
// All GEMM FLOPs via v_wmma_f32_16x16x32_bf16 (wave32 WMMA).
// LDS tiles stored in WMMA *fragment-major* order: every fragment loads as
// 2x ds_load_b128. f32->bf16 conversion during staging via v_perm_b32 packing.
// Block tile 128x128 (8 waves = 2 mGroups x 4 nGroups; 4x2 WMMA tiles/wave).
//
// Workspace layout (floats), SD = 512*1024:
//   x:0  xn:5SD  q:10SD  attn_o:15SD  kmh:20SD(2SD) vmh:22SD(2SD)
//   kg:24SD(4SD) vg:28SD(4SD) scores:32SD(12SD)  ffg:32SD(alias) ffu:46SD(14SD)
// Total = 60*SD floats = ~126 MB.
// ============================================================================

typedef __attribute__((ext_vector_type(16))) __bf16        v16bf;
typedef __attribute__((ext_vector_type(8)))  float         v8f;
typedef __attribute__((ext_vector_type(4)))  float         v4f;
typedef __attribute__((ext_vector_type(4)))  unsigned int  v4u;
typedef __attribute__((ext_vector_type(2)))  unsigned int  v2u;

union FragBF { v16bf v; v4u q[2]; };

__device__ __forceinline__ unsigned short f2bf(float f) {
  union { float f; unsigned int u; } c; c.f = f;
  return (unsigned short)((c.u + 0x8000u) >> 16);          // round-half-up
}
// Pack two f32 -> two bf16 in one dword: bytes {hi[3],hi[2],lo[3],lo[2]}
__device__ __forceinline__ unsigned int pack2(float lo, float hi) {
  union { float f; unsigned int u; } a, b; a.f = lo; b.f = hi;
  return __builtin_amdgcn_perm(b.u + 0x8000u, a.u + 0x8000u, 0x07060302u);
}

// ----------------------------------------------------------------------------
// Batched GEMM: C[M,N] (+)= A[M,K] * op(B),  op(B)=B[K,N] or B[N,K]^T (transB).
// Grid: (N/128, M/128, batch). Block: 256 threads = 8 waves.
// LDS fragment-major layout:
//   AsF: 8 mTiles x 32 lanes x 16 bf16 (lane = m + 16*half; slot s: K=16*half+s)
//        -> fragment load = 2x ds_load_b128 per tile.
//   BsF: 8 nTiles x 32 lanes x 16 bf16 (lane = n + 16*(k>>4); slot = k&15).
// Wave w: mG = w&1 (rows 64*mG..), nG = w>>1 (cols 32*nG..): 4x2 WMMA tiles.
// ----------------------------------------------------------------------------
__global__ void gemm_bf16_wmma(const float* __restrict__ A, int lda, long long sA,
                               const float* __restrict__ B, int ldb, long long sB,
                               float* __restrict__ C, int ldc, long long sC,
                               int K, int transB, int accum) {
  A += (long long)blockIdx.z * sA;
  B += (long long)blockIdx.z * sB;
  C += (long long)blockIdx.z * sC;
  const int mBase = blockIdx.y * 128;
  const int nBase = blockIdx.x * 128;

  __shared__ unsigned int AsF[8 * 32 * 8];   // 128x32 bf16, fragment-major (8KB)
  __shared__ unsigned int BsF[8 * 32 * 8];   //  32x128 bf16, fragment-major (8KB)

  const int tid  = threadIdx.x;
  const int w    = tid >> 5;
  const int lane = tid & 31;
  const int mG   = w & 1;              // wave M-group: 4 tiles, rows 64*mG..
  const int nG   = w >> 1;             // wave N-group: 2 tiles, cols 32*nG..
  const int ln15 = lane & 15;
  const int hl   = lane >> 4;

  v8f acc[4][2];
  #pragma unroll
  for (int mi = 0; mi < 4; ++mi)
    #pragma unroll
    for (int ni = 0; ni < 2; ++ni)
      acc[mi][ni] = (v8f){0.f,0.f,0.f,0.f,0.f,0.f,0.f,0.f};

  for (int kb = 0; kb < K; kb += 32) {
    __syncthreads();                   // protect previous-iteration LDS reads
    // ---- stage A tile (128x32): 1024 float4 quads, 4 per thread
    #pragma unroll
    for (int i = 0; i < 4; ++i) {
      int gi = tid + 256 * i;          // 0..1023
      int r  = gi >> 3;                // row 0..127
      int c4 = (gi & 7) * 4;           // K offset 0,4,...,28
      v4f f = *(const v4f*)(A + (size_t)(mBase + r) * lda + (kb + c4));
      int mT = r >> 4, m = r & 15;
      int hf = c4 >> 4;
      int vv = (c4 & 15) >> 1;
      v2u p = { pack2(f[0], f[1]), pack2(f[2], f[3]) };
      *(v2u*)&AsF[(size_t)((mT * 32) + m + 16 * hf) * 8 + vv] = p;
    }
    // ---- stage B tile (32x128): 1024 quads, 4 per thread
    if (transB) {                      // B[n][k], contiguous along K
      #pragma unroll
      for (int i = 0; i < 4; ++i) {
        int gi = tid + 256 * i;        // 0..1023
        int n  = gi >> 3;              // 0..127
        int k4 = (gi & 7) * 4;
        v4f f = *(const v4f*)(B + (size_t)(nBase + n) * ldb + (kb + k4));
        int nT = n >> 4, n15 = n & 15;
        int hf = k4 >> 4;
        v2u p = { pack2(f[0], f[1]), pack2(f[2], f[3]) };
        *(v2u*)&BsF[(size_t)((nT * 32) + n15 + 16 * hf) * 8 + ((k4 & 15) >> 1)] = p;
      }
    } else {                           // B[k][n], contiguous along N
      unsigned short* bs = (unsigned short*)BsF;
      #pragma unroll
      for (int i = 0; i < 4; ++i) {
        int gi = tid + 256 * i;        // 0..1023
        int k  = gi >> 5;              // 0..31
        int n4 = (gi & 31) * 4;        // 0..124
        v4f f = *(const v4f*)(B + (size_t)(kb + k) * ldb + (nBase + n4));
        int hf = k >> 4, slot = k & 15;
        #pragma unroll
        for (int j = 0; j < 4; ++j) {
          int n = n4 + j;
          bs[(size_t)(((n >> 4) * 32) + (n & 15) + 16 * hf) * 16 + slot] = f2bf(f[j]);
        }
      }
    }
    if (kb + 32 < K)                   // speculative prefetch of next A tile
      __builtin_prefetch(A + (size_t)(mBase + tid) * lda + kb + 32, 0, 1);
    __syncthreads();

    // ---- fragment loads (2x ds_load_b128 each) + 8 WMMAs
    FragBF a[4], b[2];
    #pragma unroll
    for (int t = 0; t < 4; ++t) {
      const v4u* p = (const v4u*)&AsF[(size_t)(((mG * 4 + t) * 32) + lane) * 8];
      a[t].q[0] = p[0]; a[t].q[1] = p[1];
    }
    #pragma unroll
    for (int t = 0; t < 2; ++t) {
      const v4u* p = (const v4u*)&BsF[(size_t)(((nG * 2 + t) * 32) + lane) * 8];
      b[t].q[0] = p[0]; b[t].q[1] = p[1];
    }
    #pragma unroll
    for (int mi = 0; mi < 4; ++mi)
      #pragma unroll
      for (int ni = 0; ni < 2; ++ni)
        acc[mi][ni] = __builtin_amdgcn_wmma_f32_16x16x32_bf16(
            false, a[mi].v, false, b[ni].v, (short)0, acc[mi][ni], false, false);
  }

  // ---- store C (ISA C/D layout: VGPR j -> row 16*mT + 8*half + j, col lane&15)
  #pragma unroll
  for (int mi = 0; mi < 4; ++mi) {
    #pragma unroll
    for (int ni = 0; ni < 2; ++ni) {
      int rb = mBase + (mG * 4 + mi) * 16 + hl * 8;
      int cc = nBase + (nG * 2 + ni) * 16 + ln15;
      #pragma unroll
      for (int j = 0; j < 8; ++j) {
        size_t idx = (size_t)(rb + j) * ldc + cc;
        if (accum) C[idx] += acc[mi][ni][j];
        else       C[idx]  = acc[mi][ni][j];
      }
    }
  }
}

// ----------------------------------------------------------------------------
// Elementwise / reduction kernels
// ----------------------------------------------------------------------------
static constexpr long long SDc = 512LL * 1024LL;

__global__ void init_x_kernel(const int* __restrict__ ids,
                              const float* __restrict__ embed,
                              const float* __restrict__ ig, const float* __restrict__ fg,
                              const float* __restrict__ og, const float* __restrict__ gg,
                              float* __restrict__ x) {
  size_t i = (size_t)blockIdx.x * 256 + threadIdx.x;     // 5*SD threads
  int stream = (int)(i / SDc);
  size_t r = i % SDc;
  if (stream == 0) {
    int row = (int)(r >> 10), col = (int)(r & 1023);
    x[i] = embed[(size_t)ids[row] * 1024 + col];
  } else {
    const float* src = (stream == 1) ? ig : (stream == 2) ? fg : (stream == 3) ? og : gg;
    x[i] = src[r];
  }
}

__global__ void gating_kernel(const float* __restrict__ x, const float* __restrict__ cells,
                              const float* __restrict__ ib, const float* __restrict__ fb,
                              const float* __restrict__ ob, const float* __restrict__ gb,
                              float* __restrict__ oc, float* __restrict__ os) {
  size_t i = (size_t)blockIdx.x * 256 + threadIdx.x;     // SD threads
  float I = 1.f / (1.f + expf(-(x[SDc + i]     + ib[i])));
  float F = 1.f / (1.f + expf(-(x[2 * SDc + i] + fb[i])));
  float O = 1.f / (1.f + expf(-(x[3 * SDc + i] + ob[i])));
  float G = tanhf(x[4 * SDc + i] + gb[i]);
  float nc = I * G + F * cells[i];
  oc[i] = nc;
  os[i] = O * tanhf(nc);
}

__global__ void rmsnorm_kernel(const float* __restrict__ x, const float* __restrict__ w,
                               float* __restrict__ xn) {
  __shared__ float red[256];
  int row = blockIdx.x, t = threadIdx.x;
  const float* xr = x + (size_t)row * 1024;
  v4f acc4 = *(const v4f*)(xr + t * 4);                  // 1024 = 256 * 4
  float s = acc4[0]*acc4[0] + acc4[1]*acc4[1] + acc4[2]*acc4[2] + acc4[3]*acc4[3];
  red[t] = s; __syncthreads();
  for (int o = 128; o > 0; o >>= 1) { if (t < o) red[t] += red[t + o]; __syncthreads(); }
  float r = rsqrtf(red[0] * (1.f / 1024.f) + 1e-5f);
  v4f wv = *(const v4f*)(w + t * 4);
  v4f ov = { acc4[0]*r*wv[0], acc4[1]*r*wv[1], acc4[2]*r*wv[2], acc4[3]*r*wv[3] };
  *(v4f*)(xn + (size_t)row * 1024 + t * 4) = ov;
}

__global__ void rope_kernel(float* __restrict__ buf, int posBase, int posMod) {
  int row = blockIdx.x;
  float pos = (float)(posBase + (row % posMod));
  float* xr = buf + (size_t)row * 1024;
  for (int p = threadIdx.x; p < 512; p += 256) {     // 8 heads x 64 pairs
    int h = p >> 6, i = p & 63;
    int d0 = h * 128 + i, d1 = d0 + 64;
    float inv = powf(10000.f, -(float)(2 * i) * (1.f / 128.f));
    float f = pos * inv;
    float c = cosf(f), s = sinf(f);
    float x1 = xr[d0], x2 = xr[d1];
    xr[d0] = x1 * c - x2 * s;
    xr[d1] = x2 * c + x1 * s;
  }
}

__device__ __forceinline__ bool key_visible(int j, int q) {
  if (j < 512) return true;                 // memory keys, kpos 0..511 <= qpos
  if (j < 1024) return (j - 512) <= q;      // hidden keys, kpos 512+j'
  return (j - 1024) <= q;                   // gate keys share qpos
}

__global__ void softmax_kernel(float* __restrict__ sc, int nk) {
  int q = blockIdx.x, h = blockIdx.y, t = threadIdx.x;
  float* row = sc + ((size_t)h * 512 + q) * nk;
  __shared__ float red[256];
  const float scale = 0.0883883476483184f;  // 1/sqrt(128)
  float vloc[6];                            // nk <= 1536 = 6*256
  int nloc = nk >> 8;
  float m = -1e30f;
  #pragma unroll
  for (int c = 0; c < 6; ++c) {
    if (c >= nloc) break;
    int j = t + c * 256;
    float v = key_visible(j, q) ? row[j] * scale : -1e30f;
    vloc[c] = v;
    m = fmaxf(m, v);
  }
  red[t] = m; __syncthreads();
  for (int o = 128; o > 0; o >>= 1) { if (t < o) red[t] = fmaxf(red[t], red[t + o]); __syncthreads(); }
  m = red[0]; __syncthreads();
  float s = 0.f;
  #pragma unroll
  for (int c = 0; c < 6; ++c) {
    if (c >= nloc) break;
    float e = (vloc[c] > -1e29f) ? expf(vloc[c] - m) : 0.f;
    vloc[c] = e;
    s += e;
  }
  red[t] = s; __syncthreads();
  for (int o = 128; o > 0; o >>= 1) { if (t < o) red[t] += red[t + o]; __syncthreads(); }
  float inv = 1.f / red[0];
  #pragma unroll
  for (int c = 0; c < 6; ++c) {
    if (c >= nloc) break;
    row[t + c * 256] = vloc[c] * inv;
  }
}

__global__ void swiglu_kernel(float* __restrict__ g, const float* __restrict__ u) {
  size_t i = (size_t)blockIdx.x * 256 + threadIdx.x;
  float gv = g[i];
  g[i] = (gv / (1.f + expf(-gv))) * u[i];
}

// ----------------------------------------------------------------------------
// Host orchestration
// ----------------------------------------------------------------------------
static inline void gemm(hipStream_t s, const float* A, int lda, long long sA,
                        const float* B, int ldb, long long sB,
                        float* C, int ldc, long long sC,
                        int M, int N, int K, int transB, int accum, int batch) {
  dim3 grid(N / 128, M / 128, batch);
  gemm_bf16_wmma<<<grid, dim3(256), 0, s>>>(A, lda, sA, B, ldb, sB, C, ldc, sC,
                                            K, transB, accum);
}

extern "C" void kernel_launch(void* const* d_in, const int* in_sizes, int n_in,
                              void* d_out, int out_size, void* d_ws, size_t ws_size,
                              hipStream_t stream) {
  (void)in_sizes; (void)n_in; (void)out_size; (void)ws_size;
  const long long SD = 512LL * 1024LL;
  const long long DD = 1024LL * 1024LL;
  const long long DF = 1024LL * 2816LL;
  const long long FD = 2816LL * 1024LL;
  const long long SF = 2560LL * 2816LL;

  const int*   ids    = (const int*)  d_in[0];
  const float* cells  = (const float*)d_in[1];
  const float* state  = (const float*)d_in[2];
  const float* i_gt   = (const float*)d_in[3];
  const float* f_gt   = (const float*)d_in[4];
  const float* o_gt   = (const float*)d_in[5];
  const float* g_gt   = (const float*)d_in[6];
  const float* embed  = (const float*)d_in[7];
  const float* ln1    = (const float*)d_in[8];
  const float* ln2    = (const float*)d_in[9];
  const float* Wq     = (const float*)d_in[10];
  const float* Wk     = (const float*)d_in[11];
  const float* Wv     = (const float*)d_in[12];
  const float* Wo     = (const float*)d_in[13];
  const float* mWk    = (const float*)d_in[14];
  const float* mWv    = (const float*)d_in[15];
  const float* gWq    = (const float*)d_in[16];
  const float* gWk    = (const float*)d_in[17];
  const float* gWv    = (const float*)d_in[18];
  const float* Wg     = (const float*)d_in[19];
  const float* Wu     = (const float*)d_in[20];
  const float* Wd     = (const float*)d_in[21];
  const float* i_b    = (const float*)d_in[22];
  const float* f_b    = (const float*)d_in[23];
  const float* o_b    = (const float*)d_in[24];
  const float* g_b    = (const float*)d_in[25];
  float* out = (float*)d_out;                 // [cells(4SD) | state(4SD)]
  float* ws  = (float*)d_ws;

  float* x      = ws;
  float* xn     = ws + 5 * SD;
  float* q      = ws + 10 * SD;
  float* attn_o = ws + 15 * SD;
  float* kmh    = ws + 20 * SD;               // 2SD: [mem K | hidden K]
  float* vmh    = ws + 22 * SD;
  float* kg     = ws + 24 * SD;               // 4SD: 4 gates
  float* vg     = ws + 28 * SD;
  float* scores = ws + 32 * SD;               // 12SD (aliases ffg)
  float* ffg    = ws + 32 * SD;               // SF
  float* ffu    = ws + 46 * SD;               // SF

  // x = [embed[ids], i_gate, f_gate, o_gate, g_gate]
  init_x_kernel<<<(unsigned)(5 * SD / 256), 256, 0, stream>>>(ids, embed, i_gt, f_gt, o_gt, g_gt, x);

  for (int l = 0; l < 4; ++l) {
    // Final gating uses gate streams AT LAYER ENTRY -> independent per layer.
    gating_kernel<<<(unsigned)(SD / 256), 256, 0, stream>>>(
        x, cells + l * SD, i_b + l * SD, f_b + l * SD, o_b + l * SD, g_b + l * SD,
        out + l * SD, out + (4 + l) * SD);

    // ---- ln1 over all 5 streams
    rmsnorm_kernel<<<2560, 256, 0, stream>>>(x, ln1 + l * 1024, xn);

    // ---- projections (all WMMA GEMMs)
    gemm(stream, xn, 1024, 0, Wq + l * DD, 1024, 0, q, 1024, 0, 512, 1024, 1024, 0, 0, 1);
    gemm(stream, xn + SD, 1024, SD, gWq + l * 4 * DD, 1024, DD, q + SD, 1024, SD,
         512, 1024, 1024, 0, 0, 4);
    gemm(stream, state + l * SD, 1024, 0, mWk + l * DD, 1024, 0, kmh, 1024, 0, 512, 1024, 1024, 0, 0, 1);
    gemm(stream, state + l * SD, 1024, 0, mWv + l * DD, 1024, 0, vmh, 1024, 0, 512, 1024, 1024, 0, 0, 1);
    gemm(stream, xn, 1024, 0, Wk + l * DD, 1024, 0, kmh + SD, 1024, 0, 512, 1024, 1024, 0, 0, 1);
    gemm(stream, xn, 1024, 0, Wv + l * DD, 1024, 0, vmh + SD, 1024, 0, 512, 1024, 1024, 0, 0, 1);
    gemm(stream, xn + SD, 1024, SD, gWk + l * 4 * DD, 1024, DD, kg, 1024, SD, 512, 1024, 1024, 0, 0, 4);
    gemm(stream, xn + SD, 1024, SD, gWv + l * 4 * DD, 1024, DD, vg, 1024, SD, 512, 1024, 1024, 0, 0, 4);

    // ---- RoPE (Q: qpos 512..1023; K_mem+K_hid: 0..1023; gate K: 512..1023)
    rope_kernel<<<2560, 256, 0, stream>>>(q, 512, 512);
    rope_kernel<<<1024, 256, 0, stream>>>(kmh, 0, 1024);
    rope_kernel<<<2048, 256, 0, stream>>>(kg, 512, 512);

    // ---- hidden-stream attention (batched over 8 heads)
    gemm(stream, q, 1024, 128, kmh, 1024, 128, scores, 1024, 512LL * 1024,
         512, 1024, 128, 1, 0, 8);                       // S = Q K^T
    softmax_kernel<<<dim3(512, 8), 256, 0, stream>>>(scores, 1024);
    gemm(stream, scores, 1024, 512LL * 1024, vmh, 1024, 128, attn_o, 1024, 128,
         512, 128, 1024, 0, 0, 8);                       // O = P V

    // ---- gate-stream attention (keys = mem+hidden (1024) ++ own gate (512))
    for (int g = 0; g < 4; ++g) {
      float* qg = q + (1 + g) * SD;
      gemm(stream, qg, 1024, 128, kmh, 1024, 128, scores, 1536, 512LL * 1536,
           512, 1024, 128, 1, 0, 8);
      gemm(stream, qg, 1024, 128, kg + g * SD, 1024, 128, scores + 1024, 1536, 512LL * 1536,
           512, 512, 128, 1, 0, 8);
      softmax_kernel<<<dim3(512, 8), 256, 0, stream>>>(scores, 1536);
      float* og2 = attn_o + (1 + g) * SD;
      gemm(stream, scores, 1536, 512LL * 1536, vmh, 1024, 128, og2, 1024, 128,
           512, 128, 1024, 0, 0, 8);
      gemm(stream, scores + 1024, 1536, 512LL * 1536, vg + g * SD, 1024, 128, og2, 1024, 128,
           512, 128, 512, 0, 1, 8);                      // accumulate gate-key part
    }

    // ---- output projection fused with residual: x += attn_o @ Wo[l]
    gemm(stream, attn_o, 1024, 0, Wo + l * DD, 1024, 0, x, 1024, 0, 2560, 1024, 1024, 0, 1, 1);

    // ---- FFN (SwiGLU), down-proj fused with residual
    rmsnorm_kernel<<<2560, 256, 0, stream>>>(x, ln2 + l * 1024, xn);
    gemm(stream, xn, 1024, 0, Wg + l * DF, 2816, 0, ffg, 2816, 0, 2560, 2816, 1024, 0, 0, 1);
    gemm(stream, xn, 1024, 0, Wu + l * DF, 2816, 0, ffu, 2816, 0, 2560, 2816, 1024, 0, 0, 1);
    swiglu_kernel<<<(unsigned)(SF / 256), 256, 0, stream>>>(ffg, ffu);
    gemm(stream, ffg, 2816, 0, Wd + l * FD, 1024, 0, x, 1024, 0, 2560, 1024, 2816, 0, 1, 1);
  }
}
